// RingDilatedAttentionHilbertOptimizedCorrect_6914897346643
// MI455X (gfx1250) — matrix operand order repaired and checked by reading, compiled, not run on hardware
//
#include <hip/hip_runtime.h>
#include <stdint.h>
#include <stddef.h>

// ---------------------------------------------------------------------------
// MI455X (gfx1250): bf16 WMMA everywhere, double-buffered async loads to LDS.
// ---------------------------------------------------------------------------

typedef __bf16 bf16;
typedef __attribute__((ext_vector_type(16))) __bf16 v16bf;
typedef __attribute__((ext_vector_type(4)))  __bf16 v4bf;
typedef __attribute__((ext_vector_type(8)))  float  v8f;
typedef __attribute__((ext_vector_type(4)))  int    v4i;

#define AS1 __attribute__((address_space(1)))
#define AS3 __attribute__((address_space(3)))

#if defined(__HIP_DEVICE_COMPILE__) && __has_builtin(__builtin_amdgcn_global_load_async_to_lds_b128)
#define HAVE_ASYNC_LDS 1
#else
#define HAVE_ASYNC_LDS 0
#endif

// 16-byte global -> LDS copy. Per-lane addresses (one wave instruction per call).
__device__ __forceinline__ void cp_async16(void* lds_dst, const void* g_src) {
#if HAVE_ASYNC_LDS
  __builtin_amdgcn_global_load_async_to_lds_b128((AS1 v4i*)g_src, (AS3 v4i*)lds_dst, 0, 0);
#else
  *(uint4*)lds_dst = *(const uint4*)g_src;
#endif
}

// Wait until at most N async ops remain outstanding (in-order retirement).
template <int N>
__device__ __forceinline__ void async_wait_n() {
#if HAVE_ASYNC_LDS
#if __has_builtin(__builtin_amdgcn_s_wait_asynccnt)
  __builtin_amdgcn_s_wait_asynccnt(N);
#else
  asm volatile("s_wait_asynccnt %0" ::"i"(N) : "memory");
#endif
#endif
}

__device__ __forceinline__ v8f wmma_bf16(v16bf a, v16bf b, v8f c) {
  // D(16x16,f32) = A(16x32,bf16) x B(32x16,bf16) + C
  return __builtin_amdgcn_wmma_f32_16x16x32_bf16(false, a, false, b, (short)0, c,
                                                 false, false);
}

// A fragment (16 rows x 32 K), LDS tile [row][k], row stride ld.
// Lane m=lane&15 holds row m; K = e + 16*hi  (contiguous 16 K per lane half).
__device__ __forceinline__ v16bf load_frag_a(const bf16* p, int ld) {
  const int lane = threadIdx.x & 31;
  const int m = lane & 15;
  const int hi = lane >> 4;
  v16bf f;
#pragma unroll
  for (int e = 0; e < 16; ++e) {
    const int k = e + 16 * hi;
    f[e] = p[m * ld + k];
  }
  return f;
}

// B fragment (32 K x 16 N) from LDS tile stored [n][k] (k contiguous).
__device__ __forceinline__ v16bf load_frag_b_nmajor(const bf16* p, int ld) {
  const int lane = threadIdx.x & 31;
  const int n = lane & 15;
  const int hi = lane >> 4;
  v16bf f;
#pragma unroll
  for (int e = 0; e < 16; ++e) f[e] = p[n * ld + (e + 16 * hi)];
  return f;
}

// B fragment (32 K x 16 N) from LDS tile stored [k][n] (n contiguous).
__device__ __forceinline__ v16bf load_frag_b_kmajor(const bf16* p, int ld) {
  const int lane = threadIdx.x & 31;
  const int n = lane & 15;
  const int hi = lane >> 4;
  v16bf f;
#pragma unroll
  for (int e = 0; e < 16; ++e) f[e] = p[(e + 16 * hi) * ld + n];
  return f;
}

// wave32: reduce across the 16 lanes sharing one C-matrix row half.
__device__ __forceinline__ float rowmax16(float v) {
  v = fmaxf(v, __shfl_xor(v, 1, 32));
  v = fmaxf(v, __shfl_xor(v, 2, 32));
  v = fmaxf(v, __shfl_xor(v, 4, 32));
  v = fmaxf(v, __shfl_xor(v, 8, 32));
  return v;
}
__device__ __forceinline__ float rowsum16(float v) {
  v += __shfl_xor(v, 1, 32);
  v += __shfl_xor(v, 2, 32);
  v += __shfl_xor(v, 4, 32);
  v += __shfl_xor(v, 8, 32);
  return v;
}

// ---------------------------------------------------------------------------
// Hilbert permutation (n = 64 grid for every segment length here).
// ---------------------------------------------------------------------------
__device__ int hilbert_lin(int d) {
  int x = 0, y = 0, t = d;
  for (int s = 1; s < 64; s <<= 1) {
    const int rx = 1 & (t >> 1);
    const int ry = 1 & (t ^ rx);
    if (ry == 0) {
      if (rx == 1) { x = s - 1 - x; y = s - 1 - y; }
      const int tmp = x; x = y; y = tmp;
    }
    x += s * rx;
    y += s * ry;
    t >>= 2;
  }
  return y * 64 + x;
}

__global__ void build_perms(int* __restrict__ p2048, int* __restrict__ p4096) {
  const int tid = blockIdx.x * blockDim.x + threadIdx.x;
  if (tid < 4096) p4096[tid] = hilbert_lin(tid);   // L=4096: no filtering
  if (tid == 0) {                                  // L=2048: needs compaction
    int cnt = 0;
    for (int d = 0; d < 4096; ++d) {
      const int lin = hilbert_lin(d);
      if (lin < 2048) p2048[cnt++] = lin;
    }
  }
}

// ---------------------------------------------------------------------------
// f32 -> bf16 conversion (vectorized x4)
// ---------------------------------------------------------------------------
__global__ void cvt_bf16(const float* __restrict__ in, bf16* __restrict__ out, int n4) {
  const int i = blockIdx.x * blockDim.x + threadIdx.x;
  if (i < n4) {
    const float4 v = ((const float4*)in)[i];
    v4bf o;
    o[0] = (bf16)v.x; o[1] = (bf16)v.y; o[2] = (bf16)v.z; o[3] = (bf16)v.w;
    ((v4bf*)out)[i] = o;
  }
}

// ---------------------------------------------------------------------------
// NT GEMM: C[M,N] = A[M,K] * B[N,K]^T + bias   (A,B bf16 row-major, K contig)
// Block tile 128x64, 8 waves; wave owns 32x32 (2x2 WMMA frags).
// Double-buffered LDS, per-thread fixed staging slots (pointer += 64B / step).
// ---------------------------------------------------------------------------
#define GEMM_LDT 56  // 112B row stride: 16B-aligned, <=2-way bank conflicts

template <bool OUT_BF16>
__global__ __launch_bounds__(256) void gemm_bf16_nt(
    const bf16* __restrict__ A, const bf16* __restrict__ B,
    const float* __restrict__ bias, void* __restrict__ Cout,
    int M, int N, int K) {
  __shared__ __align__(16) bf16 As[2][128 * GEMM_LDT];
  __shared__ __align__(16) bf16 Bs[2][64 * GEMM_LDT];
  const int tid = threadIdx.x;
  const int wave = tid >> 5;
  const int bm = blockIdx.x, bn = blockIdx.y;
  const int wr = (wave >> 1) * 32;  // 0..96
  const int wc = (wave & 1) * 32;   // 0,32

  // fixed staging slots: A = 128 rows x 32 elems (2 chunks/thread),
  //                      B =  64 rows x 32 elems (1 chunk/thread)
  const int rowA = tid >> 1, eoA = (tid & 1) * 16;
  const int rowB = tid >> 2, eoB = (tid & 3) * 8;
  const bf16* gA = A + (size_t)(bm * 128 + rowA) * K + eoA;
  const bf16* gB = B + (size_t)(bn * 64 + rowB) * K + eoB;
  bf16* lA0 = &As[0][rowA * GEMM_LDT + eoA];
  bf16* lA1 = &As[1][rowA * GEMM_LDT + eoA];
  bf16* lB0 = &Bs[0][rowB * GEMM_LDT + eoB];
  bf16* lB1 = &Bs[1][rowB * GEMM_LDT + eoB];

  // prologue: stage k-step 0 into buffer 0 (3 async ops / wave)
  cp_async16(lA0, gA);
  cp_async16(lA0 + 8, gA + 8);
  cp_async16(lB0, gB);
  gA += 32; gB += 32;

  v8f acc[2][2] = {};
  const int nsteps = K >> 5;
  for (int s = 0; s < nsteps; ++s) {
    const int cur = s & 1;
    if (s + 1 < nsteps) {  // issue next step into the other buffer
      bf16* nA = cur ? lA0 : lA1;
      bf16* nB = cur ? lB0 : lB1;
      cp_async16(nA, gA);
      cp_async16(nA + 8, gA + 8);
      cp_async16(nB, gB);
      gA += 32; gB += 32;
      async_wait_n<3>();  // current step's batch has retired (in-order)
    } else {
      async_wait_n<0>();
    }
    __syncthreads();

    const v16bf a0 = load_frag_a(&As[cur][wr * GEMM_LDT], GEMM_LDT);
    const v16bf a1 = load_frag_a(&As[cur][(wr + 16) * GEMM_LDT], GEMM_LDT);
    const v16bf b0 = load_frag_b_nmajor(&Bs[cur][wc * GEMM_LDT], GEMM_LDT);
    const v16bf b1 = load_frag_b_nmajor(&Bs[cur][(wc + 16) * GEMM_LDT], GEMM_LDT);
    acc[0][0] = wmma_bf16(a0, b0, acc[0][0]);
    acc[0][1] = wmma_bf16(a0, b1, acc[0][1]);
    acc[1][0] = wmma_bf16(a1, b0, acc[1][0]);
    acc[1][1] = wmma_bf16(a1, b1, acc[1][1]);
    __syncthreads();  // releases buffer 'cur' for overwrite at step s+2
  }

  const int lane = tid & 31, nl = lane & 15, hi = lane >> 4;
#pragma unroll
  for (int mf = 0; mf < 2; ++mf)
#pragma unroll
    for (int nf = 0; nf < 2; ++nf) {
      const int col = bn * 64 + wc + nf * 16 + nl;
      const float bv = bias[col];
#pragma unroll
      for (int r = 0; r < 8; ++r) {
        const int row = bm * 128 + wr + mf * 16 + r + 8 * hi;
        const float v = acc[mf][nf][r] + bv;
        if (OUT_BF16) ((bf16*)Cout)[(size_t)row * N + col] = (bf16)v;
        else          ((float*)Cout)[(size_t)row * N + col] = v;
      }
    }
}

// ---------------------------------------------------------------------------
// Dilated Hilbert attention, flash-style.
// Grid: x = 128 q-tiles (seg0:32, seg1:32, seg2:64), y = 16 heads.
// Block: 128 threads (4 waves); wave w owns q rows [w*16, w*16+16).
// qkv layout: [8192][3072] bf16, row = q(0..1023) | k(1024..2047) | v(2048..)
// K/V chunks (64 keys x 64 dims) double-buffered through async-to-LDS.
// ---------------------------------------------------------------------------
#define LDQ 72  // 144B row stride: 16B-aligned, 2-way bank conflicts max

__global__ __launch_bounds__(128) void dil_attn(
    const bf16* __restrict__ qkv, const int* __restrict__ p2048,
    const int* __restrict__ p4096, bf16* __restrict__ attn) {
  __shared__ __align__(16) bf16 Qs[64 * LDQ];
  __shared__ __align__(16) bf16 Ks[2][64 * LDQ];
  __shared__ __align__(16) bf16 Vs[2][64 * LDQ];
  __shared__ __align__(16) bf16 Ps[4 * 16 * LDQ];  // per-wave P transpose scratch

  const int bx = blockIdx.x, head = blockIdx.y;
  int seg_start, dil, kd_len, qt;
  const int* perm;
  if (bx < 32)      { seg_start = 0;    dil = 1; kd_len = 2048; qt = bx;      perm = p2048; }
  else if (bx < 64) { seg_start = 2048; dil = 2; kd_len = 1024; qt = bx - 32; perm = p2048; }
  else              { seg_start = 4096; dil = 4; kd_len = 1024; qt = bx - 64; perm = p4096; }

  const int tid = threadIdx.x;
  const int wave = tid >> 5;
  const int lane = tid & 31, nl = lane & 15, hi = lane >> 4;

  // staging slots: 2 threads per row, 32 elems (4x16B chunks) per thread
  const int rowS = tid >> 1;
  const int eo = (tid & 1) * 32;

  // ---- stage gathered Q tile (64 rows x 64 dims), 4 async ops / wave ----
  {
    const int g = seg_start + perm[qt * 64 + rowS];
    const bf16* src = qkv + (size_t)g * 3072 + head * 64 + eo;
    bf16* dst = &Qs[rowS * LDQ + eo];
    cp_async16(dst, src);
    cp_async16(dst + 8, src + 8);
    cp_async16(dst + 16, src + 16);
    cp_async16(dst + 24, src + 24);
  }

  // per-thread K/V staging pointers (perm walked by 64*dil per chunk)
  const int* pK = perm + rowS * dil;
  bf16* dK[2] = { &Ks[0][rowS * LDQ + eo], &Ks[1][rowS * LDQ + eo] };
  bf16* dV[2] = { &Vs[0][rowS * LDQ + eo], &Vs[1][rowS * LDQ + eo] };
  const int nch = kd_len >> 6;

  // prologue: stage K/V chunk 0 into buffer 0 (8 async ops / wave)
  {
    const int g = seg_start + pK[0];
    pK += 64 * dil;
    const bf16* sK = qkv + (size_t)g * 3072 + 1024 + head * 64 + eo;
    const bf16* sV = sK + 1024;
#pragma unroll
    for (int c = 0; c < 4; ++c) cp_async16(dK[0] + c * 8, sK + c * 8);
#pragma unroll
    for (int c = 0; c < 4; ++c) cp_async16(dV[0] + c * 8, sV + c * 8);
  }

  async_wait_n<8>();  // Q (first 4 ops) has retired; chunk 0 may be in flight
  __syncthreads();

  const v16bf qa0 = load_frag_a(&Qs[(wave * 16) * LDQ], LDQ);       // kdim 0..31
  const v16bf qa1 = load_frag_a(&Qs[(wave * 16) * LDQ + 32], LDQ);  // kdim 32..63

  float m_i[8], l_i[8];
  v8f acc[4] = {};
#pragma unroll
  for (int r = 0; r < 8; ++r) { m_i[r] = -1e30f; l_i[r] = 0.f; }

  for (int kc = 0; kc < nch; ++kc) {
    const int cur = kc & 1;
    if (kc + 1 < nch) {  // issue next chunk into the other buffer
      const int g = seg_start + pK[0];
      pK += 64 * dil;
      const bf16* sK = qkv + (size_t)g * 3072 + 1024 + head * 64 + eo;
      const bf16* sV = sK + 1024;
#pragma unroll
      for (int c = 0; c < 4; ++c) cp_async16(dK[cur ^ 1] + c * 8, sK + c * 8);
#pragma unroll
      for (int c = 0; c < 4; ++c) cp_async16(dV[cur ^ 1] + c * 8, sV + c * 8);
      async_wait_n<8>();  // current chunk retired; next chunk outstanding
    } else {
      async_wait_n<0>();
    }
    __syncthreads();

    const bf16* Kb = &Ks[cur][0];
    const bf16* Vb = &Vs[cur][0];

    // ---- S = Q * K^T (16q x 64k per wave) ----
    v8f s[4];
#pragma unroll
    for (int f = 0; f < 4; ++f) {
      const v16bf kb0 = load_frag_b_nmajor(&Kb[(f * 16) * LDQ], LDQ);
      const v16bf kb1 = load_frag_b_nmajor(&Kb[(f * 16) * LDQ + 32], LDQ);
      v8f z = {};
      z = wmma_bf16(qa0, kb0, z);
      s[f] = wmma_bf16(qa1, kb1, z);
    }

    // ---- online softmax (f32) ----
#pragma unroll
    for (int r = 0; r < 8; ++r) {
      float tmax = -1e30f;
#pragma unroll
      for (int f = 0; f < 4; ++f) {
        s[f][r] *= 0.125f;  // 1/sqrt(64)
        tmax = fmaxf(tmax, s[f][r]);
      }
      tmax = rowmax16(tmax);
      const float newm = fmaxf(m_i[r], tmax);
      const float corr = __expf(m_i[r] - newm);
      float psum = 0.f;
#pragma unroll
      for (int f = 0; f < 4; ++f) {
        const float p = __expf(s[f][r] - newm);
        s[f][r] = p;
        psum += p;
      }
      psum = rowsum16(psum);
      l_i[r] = l_i[r] * corr + psum;
      m_i[r] = newm;
#pragma unroll
      for (int d = 0; d < 4; ++d) acc[d][r] *= corr;
    }

    // ---- transpose P through per-wave LDS scratch (LDS in-order per wave) ----
    bf16* Pw = &Ps[(wave * 16) * LDQ];
#pragma unroll
    for (int f = 0; f < 4; ++f)
#pragma unroll
      for (int r = 0; r < 8; ++r)
        Pw[(r + 8 * hi) * LDQ + f * 16 + nl] = (bf16)s[f][r];

    // ---- acc += P(16x64) * V(64keys x 64dims) ----
#pragma unroll
    for (int kk = 0; kk < 2; ++kk) {
      const v16bf pa = load_frag_a(Pw + kk * 32, LDQ);
#pragma unroll
      for (int d = 0; d < 4; ++d) {
        const v16bf vb = load_frag_b_kmajor(&Vb[(kk * 32) * LDQ + d * 16], LDQ);
        acc[d] = wmma_bf16(pa, vb, acc[d]);
      }
    }
    __syncthreads();  // releases buffer 'cur' for overwrite at chunk kc+2
  }

  // ---- normalize and scatter output rows back through perm ----
#pragma unroll
  for (int r = 0; r < 8; ++r) {
    const int qrow = qt * 64 + wave * 16 + r + 8 * hi;
    const int g = seg_start + perm[qrow];
    const float inv_l = 1.f / l_i[r];
#pragma unroll
    for (int d = 0; d < 4; ++d)
      attn[(size_t)g * 1024 + head * 64 + d * 16 + nl] = (bf16)(acc[d][r] * inv_l);
  }
}

// ---------------------------------------------------------------------------
// Host launcher
// ---------------------------------------------------------------------------
extern "C" void kernel_launch(void* const* d_in, const int* in_sizes, int n_in,
                              void* d_out, int out_size, void* d_ws, size_t ws_size,
                              hipStream_t stream) {
  (void)in_sizes; (void)n_in; (void)out_size; (void)ws_size;
  const float* x    = (const float*)d_in[0];
  const float* Wqkv = (const float*)d_in[1];
  const float* bqkv = (const float*)d_in[2];
  const float* Wout = (const float*)d_in[3];
  const float* bout = (const float*)d_in[4];
  float* out = (float*)d_out;

  char* ws = (char*)d_ws;
  int* p2048 = (int*)ws;                 // 8 KB
  int* p4096 = (int*)(ws + 8 * 1024);    // 16 KB
  bf16* x_bf    = (bf16*)(ws + 32 * 1024);          // 16 MB
  bf16* wqkv_bf = x_bf    + (size_t)8192 * 1024;    //  6 MB
  bf16* wout_bf = wqkv_bf + (size_t)3072 * 1024;    //  2 MB
  bf16* qkv_bf  = wout_bf + (size_t)1024 * 1024;    // 48 MB
  bf16* attn_bf = qkv_bf  + (size_t)8192 * 3072;    // 16 MB

  build_perms<<<16, 256, 0, stream>>>(p2048, p4096);
  cvt_bf16<<<(8192 * 1024 / 4 + 255) / 256, 256, 0, stream>>>(x, x_bf, 8192 * 1024 / 4);
  cvt_bf16<<<(3072 * 1024 / 4 + 255) / 256, 256, 0, stream>>>(Wqkv, wqkv_bf, 3072 * 1024 / 4);
  cvt_bf16<<<(1024 * 1024 / 4 + 255) / 256, 256, 0, stream>>>(Wout, wout_bf, 1024 * 1024 / 4);

  // QKV projection: [8192,1024] x [3072,1024]^T -> bf16 qkv
  gemm_bf16_nt<true><<<dim3(64, 48), 256, 0, stream>>>(x_bf, wqkv_bf, bqkv,
                                                       (void*)qkv_bf, 8192, 3072, 1024);
  // dilated Hilbert attention -> bf16 [8192,1024]
  dil_attn<<<dim3(128, 16), 128, 0, stream>>>(qkv_bf, p2048, p4096, attn_bf);
  // output projection: [8192,1024] x [1024,1024]^T -> f32 out
  gemm_bf16_nt<false><<<dim3(64, 16), 256, 0, stream>>>(attn_bf, wout_bf, bout,
                                                        (void*)out, 8192, 1024, 1024);
}
